// MultiHeadAttention_88330297409617
// MI455X (gfx1250) — matrix operand order repaired and checked by reading, not compiled, not run on hardware
//
#include <hip/hip_runtime.h>
#include <hip/hip_bf16.h>

// ---------------------------------------------------------------------------
// MHA for MI455X (gfx1250, wave32, WMMA). bf16 16x16x32 WMMA, f32 accumulate.
// Pipeline: cvt -> Q/K/V projections (WMMA) -> scores (WMMA) -> column softmax
// stats -> fused normalize + write attn weights + A*V (WMMA) -> out proj (WMMA).
// ---------------------------------------------------------------------------

typedef __attribute__((ext_vector_type(16))) __bf16    v16bf;
typedef __attribute__((ext_vector_type(8)))  float     v8f;
typedef __attribute__((ext_vector_type(4)))  float     v4f;
typedef __attribute__((ext_vector_type(4)))  unsigned  v4u;
typedef __attribute__((ext_vector_type(2)))  unsigned  v2u;

constexpr int SEQ = 2048;
constexpr int DMODEL = 1024;
constexpr int NHEAD = 16;
constexpr int EH = 64;          // head dim
constexpr int NB = 2;           // batch
constexpr int MROWS = NB * SEQ; // 4096

union Frag16 { v4u u4[2]; unsigned u[8]; v16bf v; };
union F8     { v4f q[2];  float f[8]; };

__device__ __forceinline__ unsigned short f2bf(float f) {
    union { float f; unsigned u; } c; c.f = f;
    unsigned u = c.u;
    u += 0x7FFFu + ((u >> 16) & 1u);   // round-to-nearest-even
    return (unsigned short)(u >> 16);
}

__device__ __forceinline__ v8f wmma_bf16(v16bf a, v16bf b, v8f c) {
    return __builtin_amdgcn_wmma_f32_16x16x32_bf16(
        false, a, false, b, (short)0, c, false, false);
}

// A-fragment (16x32 bf16): row = lane&15; lanes 0-15: K 0-7 & 16-23,
// lanes 16-31: K 8-15 & 24-31 (ISA 7.12.2).
__device__ __forceinline__ v16bf load_a_bf16(const unsigned short* A, int ld,
                                             int m0, int kk) {
    int lane = threadIdx.x & 31;
    int koff = (lane & 16) ? 8 : 0;
    const unsigned short* p = A + (size_t)(m0 + (lane & 15)) * ld + kk + koff;
    Frag16 t;
    t.u4[0] = *(const v4u*)p;          // K = kk+koff   .. +7
    t.u4[1] = *(const v4u*)(p + 16);   // K = kk+16+koff.. +7
    return t.v;
}

// B-fragment (32x16 bf16) from row-major Wt[N,K] (B = Wt^T): col = lane&15,
// lanes 0-15 hold K 0-7 & 16-23, lanes 16-31 hold K 8-15 & 24-31.
__device__ __forceinline__ v16bf load_b_bf16(const unsigned short* W, int ld,
                                             int n0, int kk) {
    int lane = threadIdx.x & 31;
    int kb = (lane & 16) ? 8 : 0;
    const unsigned short* p = W + (size_t)(n0 + (lane & 15)) * ld + kk + kb;
    Frag16 t;
    t.u4[0] = *(const v4u*)p;
    t.u4[1] = *(const v4u*)(p + 16);
    return t.v;
}

// ------------------------------- cvt ---------------------------------------
__global__ void cvt_kernel(const float* __restrict__ in,
                           unsigned short* __restrict__ out, int n4) {
    int i = blockIdx.x * blockDim.x + threadIdx.x;
    if (i < n4) {
        v4f x = ((const v4f*)in)[i];
        v2u r;
        r[0] = (unsigned)f2bf(x[0]) | ((unsigned)f2bf(x[1]) << 16);
        r[1] = (unsigned)f2bf(x[2]) | ((unsigned)f2bf(x[3]) << 16);
        ((v2u*)out)[i] = r;
    }
}

// --------------------- generic C = A * W^T + bias --------------------------
// A: [4096,1024] bf16, W: [1024,1024] bf16 (row-major [N,K]).
// mode 0: bf16 out [H,B,S,E];  mode 1: bf16 out [H,B,E,S];  mode 2: f32 [M,N].
__global__ void __launch_bounds__(256)
gemm_nt_kernel(const unsigned short* __restrict__ A,
               const unsigned short* __restrict__ W,
               const float* __restrict__ bias,
               unsigned short* __restrict__ out_bf,
               float* __restrict__ out_f32, int mode) {
    constexpr int N = DMODEL, K = DMODEL;
    int wid = (blockIdx.x * blockDim.x + threadIdx.x) >> 5;
    int nt = wid & (N / 32 - 1);
    int mt = wid / (N / 32);
    int m0 = mt * 32, n0 = nt * 32;

    v8f acc[2][2] = {};
    for (int kk = 0; kk < K; kk += 32) {
        v16bf a0 = load_a_bf16(A, K, m0, kk);
        v16bf a1 = load_a_bf16(A, K, m0 + 16, kk);
        v16bf b0 = load_b_bf16(W, K, n0, kk);
        v16bf b1 = load_b_bf16(W, K, n0 + 16, kk);
        acc[0][0] = wmma_bf16(a0, b0, acc[0][0]);
        acc[0][1] = wmma_bf16(a0, b1, acc[0][1]);
        acc[1][0] = wmma_bf16(a1, b0, acc[1][0]);
        acc[1][1] = wmma_bf16(a1, b1, acc[1][1]);
    }

    int lane = threadIdx.x & 31;
    int lcol = lane & 15;
    int rbase = (lane & 16) ? 8 : 0;
#pragma unroll
    for (int mi = 0; mi < 2; ++mi)
#pragma unroll
        for (int ni = 0; ni < 2; ++ni)
#pragma unroll
            for (int r = 0; r < 8; ++r) {
                int m = m0 + 16 * mi + rbase + r;
                int n = n0 + 16 * ni + lcol;
                float v = acc[mi][ni][r] + bias[n];
                if (mode == 0) {
                    int h = n >> 6, e = n & 63, b = m >> 11, s = m & (SEQ - 1);
                    out_bf[((size_t)(h * NB + b) * SEQ + s) * EH + e] = f2bf(v);
                } else if (mode == 1) {
                    int h = n >> 6, e = n & 63, b = m >> 11, s = m & (SEQ - 1);
                    out_bf[((size_t)(h * NB + b) * EH + e) * SEQ + s] = f2bf(v);
                } else {
                    out_f32[(size_t)m * N + n] = v;
                }
            }
}

// ---------------------- scores = qh * kh^T / 8 -----------------------------
__global__ void __launch_bounds__(256)
score_kernel(const unsigned short* __restrict__ qh,
             const unsigned short* __restrict__ kh,
             float* __restrict__ attnw) {
    int wid = (blockIdx.x * blockDim.x + threadIdx.x) >> 5;
    int hb = wid >> 12;              // 4096 waves per (h,b)
    int t = wid & 4095;
    int m0 = (t >> 6) * 32, n0 = (t & 63) * 32;
    const unsigned short* Q = qh + (size_t)hb * SEQ * EH;
    const unsigned short* Kt = kh + (size_t)hb * SEQ * EH;

    v8f acc[2][2] = {};
#pragma unroll
    for (int kk = 0; kk < EH; kk += 32) {
        v16bf a0 = load_a_bf16(Q, EH, m0, kk);
        v16bf a1 = load_a_bf16(Q, EH, m0 + 16, kk);
        v16bf b0 = load_b_bf16(Kt, EH, n0, kk);
        v16bf b1 = load_b_bf16(Kt, EH, n0 + 16, kk);
        acc[0][0] = wmma_bf16(a0, b0, acc[0][0]);
        acc[0][1] = wmma_bf16(a0, b1, acc[0][1]);
        acc[1][0] = wmma_bf16(a1, b0, acc[1][0]);
        acc[1][1] = wmma_bf16(a1, b1, acc[1][1]);
    }

    float* Ob = attnw + (size_t)hb * SEQ * SEQ;
    int lane = threadIdx.x & 31;
    int lcol = lane & 15;
    int rbase = (lane & 16) ? 8 : 0;
#pragma unroll
    for (int mi = 0; mi < 2; ++mi)
#pragma unroll
        for (int ni = 0; ni < 2; ++ni)
#pragma unroll
            for (int r = 0; r < 8; ++r) {
                int m = m0 + 16 * mi + rbase + r;
                int n = n0 + 16 * ni + lcol;
                Ob[(size_t)m * SEQ + n] = acc[mi][ni][r] * 0.125f;
            }
}

// ---------------- column (query-axis) softmax statistics -------------------
__global__ void colstats_kernel(const float* __restrict__ attnw,
                                float* __restrict__ colmax,
                                float* __restrict__ colsum) {
    int col = blockIdx.x * blockDim.x + threadIdx.x;  // [0, HB*SEQ)
    int hb = col >> 11, k = col & (SEQ - 1);
    const float* p = attnw + (size_t)hb * SEQ * SEQ + k;
    float m = -INFINITY, s = 0.f;
    for (int q = 0; q < SEQ; ++q) {
        float x = p[(size_t)q * SEQ];
        float nm = fmaxf(m, x);
        s = s * __expf(m - nm) + __expf(x - nm);
        m = nm;
    }
    colmax[col] = m;
    colsum[col] = s;
}

// ------ fused: normalize weights (write back) + attn_out = W * V -----------
// One wave owns 32 q-rows x all 64 e-cols, so each score element is read,
// normalized, written back, and consumed by WMMA exactly once.
__global__ void __launch_bounds__(256)
av_kernel(float* __restrict__ attnw, const float* __restrict__ colmax,
          const float* __restrict__ colsum,
          const unsigned short* __restrict__ vh_t,   // [H,B,E,S] bf16
          unsigned short* __restrict__ concat_bf) {  // [B,S,H*E] bf16
    int wid = (blockIdx.x * blockDim.x + threadIdx.x) >> 5;
    int hb = wid >> 6;
    int q0 = (wid & 63) * 32;
    int h = hb >> 1, b = hb & 1;
    int lane = threadIdx.x & 31;
    int lrow = lane & 15;
    int koff = (lane & 16) ? 8 : 0;
    int kb = (lane & 16) ? 16 : 0;
    float* Ab = attnw + (size_t)hb * SEQ * SEQ;
    const unsigned short* Vb = vh_t + (size_t)hb * EH * SEQ;
    const float* cm = colmax + hb * SEQ;
    const float* cs = colsum + hb * SEQ;

    v8f acc[2][4] = {};
    for (int kk = 0; kk < SEQ; kk += 32) {
        F8 mlo, mhi, slo, shi;
        mlo.q[0] = *(const v4f*)(cm + kk + koff);
        mlo.q[1] = *(const v4f*)(cm + kk + koff + 4);
        mhi.q[0] = *(const v4f*)(cm + kk + 16 + koff);
        mhi.q[1] = *(const v4f*)(cm + kk + 16 + koff + 4);
        slo.q[0] = *(const v4f*)(cs + kk + koff);
        slo.q[1] = *(const v4f*)(cs + kk + koff + 4);
        shi.q[0] = *(const v4f*)(cs + kk + 16 + koff);
        shi.q[1] = *(const v4f*)(cs + kk + 16 + koff + 4);

        v16bf afrag[2];
#pragma unroll
        for (int mi = 0; mi < 2; ++mi) {
            float* prow = Ab + (size_t)(q0 + 16 * mi + lrow) * SEQ + kk;
            F8 xlo, xhi;
            xlo.q[0] = *(const v4f*)(prow + koff);
            xlo.q[1] = *(const v4f*)(prow + koff + 4);
            xhi.q[0] = *(const v4f*)(prow + 16 + koff);
            xhi.q[1] = *(const v4f*)(prow + 16 + koff + 4);
#pragma unroll
            for (int j = 0; j < 8; ++j) {
                xlo.f[j] = __expf(xlo.f[j] - mlo.f[j]) / slo.f[j];
                xhi.f[j] = __expf(xhi.f[j] - mhi.f[j]) / shi.f[j];
            }
            Frag16 pk;
#pragma unroll
            for (int j = 0; j < 4; ++j) {
                pk.u[j]     = (unsigned)f2bf(xlo.f[2 * j]) |
                              ((unsigned)f2bf(xlo.f[2 * j + 1]) << 16);
                pk.u[4 + j] = (unsigned)f2bf(xhi.f[2 * j]) |
                              ((unsigned)f2bf(xhi.f[2 * j + 1]) << 16);
            }
            afrag[mi] = pk.v;
            // final attention weights (output #2)
            *(v4f*)(prow + koff)          = xlo.q[0];
            *(v4f*)(prow + koff + 4)      = xlo.q[1];
            *(v4f*)(prow + 16 + koff)     = xhi.q[0];
            *(v4f*)(prow + 16 + koff + 4) = xhi.q[1];
        }
#pragma unroll
        for (int ni = 0; ni < 4; ++ni) {
            const unsigned short* pv =
                Vb + (size_t)(ni * 16 + lrow) * SEQ + kk + koff;
            Frag16 bt;
            bt.u4[0] = *(const v4u*)pv;
            bt.u4[1] = *(const v4u*)(pv + 16);
            v16bf bfrag = bt.v;
#pragma unroll
            for (int mi = 0; mi < 2; ++mi)
                acc[mi][ni] = wmma_bf16(afrag[mi], bfrag, acc[mi][ni]);
        }
    }

    int rbase = (lane & 16) ? 8 : 0;
#pragma unroll
    for (int mi = 0; mi < 2; ++mi)
#pragma unroll
        for (int ni = 0; ni < 4; ++ni)
#pragma unroll
            for (int r = 0; r < 8; ++r) {
                int qrow = q0 + 16 * mi + rbase + r;
                int e = ni * 16 + lrow;
                concat_bf[(size_t)(b * SEQ + qrow) * DMODEL + h * EH + e] =
                    f2bf(acc[mi][ni][r]);
            }
}

// ---------------------------------------------------------------------------
extern "C" void kernel_launch(void* const* d_in, const int* in_sizes, int n_in,
                              void* d_out, int out_size, void* d_ws,
                              size_t ws_size, hipStream_t stream) {
    const float* q    = (const float*)d_in[0];
    const float* k    = (const float*)d_in[1];
    const float* v    = (const float*)d_in[2];
    const float* Wq_w = (const float*)d_in[3];
    const float* Wq_b = (const float*)d_in[4];
    const float* Wk_w = (const float*)d_in[5];
    const float* Wk_b = (const float*)d_in[6];
    const float* Wv_w = (const float*)d_in[7];
    const float* Wv_b = (const float*)d_in[8];
    const float* Wo_w = (const float*)d_in[9];
    const float* Wo_b = (const float*)d_in[10];

    const size_t M4 = (size_t)MROWS * DMODEL;      // 4,194,304
    const size_t M1 = (size_t)DMODEL * DMODEL;     // 1,048,576

    unsigned short* ws16  = (unsigned short*)d_ws;
    unsigned short* q_bf  = ws16;
    unsigned short* k_bf  = q_bf + M4;
    unsigned short* v_bf  = k_bf + M4;
    unsigned short* wq_bf = v_bf + M4;
    unsigned short* wk_bf = wq_bf + M1;
    unsigned short* wv_bf = wk_bf + M1;
    unsigned short* wo_bf = wv_bf + M1;
    unsigned short* qh_bf = wo_bf + M1;
    unsigned short* kh_bf = qh_bf + M4;
    unsigned short* vh_t  = kh_bf + M4;
    unsigned short* cc_bf = vh_t + M4;
    float* stats  = (float*)(cc_bf + M4);          // ~64 MB into ws
    float* colmax = stats;
    float* colsum = stats + (size_t)NHEAD * NB * SEQ;

    float* outp  = (float*)d_out;                  // [B,S,1024] fp32
    float* attnw = outp + M4;                      // [H,B,S,S] fp32

    auto cvt = [&](const float* src, unsigned short* dst, size_t n) {
        int n4 = (int)(n / 4);
        cvt_kernel<<<(n4 + 255) / 256, 256, 0, stream>>>(src, dst, n4);
    };
    cvt(q, q_bf, M4);
    cvt(k, k_bf, M4);
    cvt(v, v_bf, M4);
    cvt(Wq_w, wq_bf, M1);
    cvt(Wk_w, wk_bf, M1);
    cvt(Wv_w, wv_bf, M1);
    cvt(Wo_w, wo_bf, M1);

    // projections: 4096 waves -> 512 blocks x 256 threads (8 waves)
    gemm_nt_kernel<<<512, 256, 0, stream>>>(q_bf, wq_bf, Wq_b, qh_bf, nullptr, 0);
    gemm_nt_kernel<<<512, 256, 0, stream>>>(k_bf, wk_bf, Wk_b, kh_bf, nullptr, 0);
    gemm_nt_kernel<<<512, 256, 0, stream>>>(v_bf, wv_bf, Wv_b, vh_t, nullptr, 1);

    // scores: 32 (h,b) x 4096 waves
    score_kernel<<<16384, 256, 0, stream>>>(qh_bf, kh_bf, attnw);

    // column softmax stats over the query axis
    colstats_kernel<<<(NHEAD * NB * SEQ) / 256, 256, 0, stream>>>(attnw, colmax,
                                                                 colsum);

    // fused normalize + weight writeback + A*V: 2048 waves
    av_kernel<<<256, 256, 0, stream>>>(attnw, colmax, colsum, vh_t, cc_bf);

    // output projection -> fp32 out
    gemm_nt_kernel<<<512, 256, 0, stream>>>(cc_bf, wo_bf, Wo_b, nullptr, outp, 2);
}
